// PureVisionMamba_6511170420827
// MI455X (gfx1250) — matrix operand order repaired and checked by reading, not compiled, not run on hardware
//
#include <hip/hip_runtime.h>
#include <hip/hip_bf16.h>

typedef __attribute__((ext_vector_type(16))) _Float16 v16h;
typedef __attribute__((ext_vector_type(8)))  _Float16 v8h;
typedef __attribute__((ext_vector_type(8)))  float    v8f;

#define BQ   8      // batch
#define EQ   256    // embed
#define LQ   256    // tokens
#define DIQ  512
#define DSQ  16
#define ROWS (BQ*LQ)   // 2048

// ---------------------------------------------------------------- utilities

__global__ void k_cvt(const float* __restrict__ s, _Float16* __restrict__ d, long n) {
  long tid = (long)blockIdx.x * blockDim.x + threadIdx.x;
  if (tid < n) d[tid] = (_Float16)s[tid];
}

// dec_wN (C,O,2,2) f32  ->  Bt (4*O rows, C cols) f16, row n = pq*O + o
__global__ void k_tdecw(const float* __restrict__ w, _Float16* __restrict__ bt, int C, int O) {
  int tid = blockIdx.x * blockDim.x + threadIdx.x;
  if (tid >= C * O * 4) return;
  int pq = tid & 3;
  int co = tid >> 2;
  int o  = co % O;
  int c  = co / O;
  bt[(size_t)(pq * O + o) * C + c] = (_Float16)w[(size_t)(c * O + o) * 4 + pq];
}

// ---------------------------------------------------------------- WMMA GEMM
// C[M,N] (f32) = A[M,K](f16 row major) * Bt[N,K](f16 row major)^T (+bias)(+add)
// one wave -> 16 x (16*NT) tile: the A fragment is reused by NT WMMAs that
// issue back-to-back after one load wait. Simple loop shape (no explicit
// double buffering) keeps live VGPRs ~90 -> no scratch spills; the scheduler
// overlaps next-iteration loads within its register budget.

__device__ __forceinline__ v16h ldA(const _Float16* __restrict__ row, int k, int half) {
  // ISA 16-bit A 16x32: e<8 -> K=k+half*8+e ; e>=8 -> K=k+16+half*8+(e-8)
  v8h a0 = *(const v8h*)(row + k + half * 8);
  v8h a1 = *(const v8h*)(row + k + 16 + half * 8);
  v16h r;
#pragma unroll
  for (int i = 0; i < 8; ++i) { r[i] = a0[i]; r[i + 8] = a1[i]; }
  return r;
}
__device__ __forceinline__ v16h ldB(const _Float16* __restrict__ row, int k, int half) {
  // ISA 16-bit B 32x16: e<8 -> K=k+half*8+e ; e>=8 -> K=k+16+half*8+(e-8), col = lane&15
  v8h b0 = *(const v8h*)(row + k + half * 8);
  v8h b1 = *(const v8h*)(row + k + 16 + half * 8);
  v16h r;
#pragma unroll
  for (int i = 0; i < 8; ++i) { r[i] = b0[i]; r[i + 8] = b1[i]; }
  return r;
}

template <int NT>
__global__ void k_gemm(const _Float16* __restrict__ A, const _Float16* __restrict__ Bt,
                       const float* __restrict__ bias, const float* __restrict__ add,
                       float* __restrict__ C, int M, int N, int K) {
  int wave = threadIdx.x >> 5;
  int lane = threadIdx.x & 31;
  int NtBlk = N / (16 * NT);
  int tile = blockIdx.x * (blockDim.x >> 5) + wave;
  if (tile >= (M >> 4) * NtBlk) return;       // wave-uniform: EXEC stays all-ones
  int mt = tile / NtBlk, nt = tile - mt * NtBlk;
  int half = lane >> 4, lm = lane & 15;

  const _Float16* aRow = A + (size_t)(mt * 16 + lm) * K;
  const _Float16* bRow[NT];
#pragma unroll
  for (int j = 0; j < NT; ++j)
    bRow[j] = Bt + (size_t)(nt * 16 * NT + j * 16 + lm) * K;

  v8f acc[NT];
#pragma unroll
  for (int j = 0; j < NT; ++j) acc[j] = (v8f){};

  for (int k = 0; k < K; k += 32) {
    if (k + 64 < K) {
      __builtin_prefetch(aRow + k + 64, 0, 3);   // global_prefetch_b8, near scope
#pragma unroll
      for (int j = 0; j < NT; ++j) __builtin_prefetch(bRow[j] + k + 64, 0, 3);
    }
    v16h av = ldA(aRow, k, half);
    v16h bv[NT];
#pragma unroll
    for (int j = 0; j < NT; ++j) bv[j] = ldB(bRow[j], k, half);
    // NT back-to-back WMMAs sharing the A fragment
#pragma unroll
    for (int j = 0; j < NT; ++j)
      acc[j] = __builtin_amdgcn_wmma_f32_16x16x32_f16(false, av, false, bv[j],
                                                      (short)0, acc[j], false, false);
  }

#pragma unroll
  for (int j = 0; j < NT; ++j) {
    int n = nt * 16 * NT + j * 16 + lm;
#pragma unroll
    for (int v = 0; v < 8; ++v) {             // C/D: row = v + 8*half, col = lane&15
      int m = mt * 16 + v + half * 8;
      float r = acc[j][v];
      if (bias) r += bias[n];
      size_t idx = (size_t)m * N + n;
      if (add)  r += add[idx];
      C[idx] = r;
    }
  }
}

// ---------------------------------------------------------------- embedding

__global__ void k_time_emb(const int* __restrict__ t,
                           const float* __restrict__ w1, const float* __restrict__ b1,
                           const float* __restrict__ w2, const float* __restrict__ b2,
                           float* __restrict__ temb) {
  __shared__ float hid[EQ];
  int b = blockIdx.x, j = threadIdx.x;
  float tf = (float)t[b];
  float xv = tf * w1[j] + b1[j];
  hid[j] = 0.5f * xv * (1.f + erff(xv * 0.70710678118654752f));   // exact gelu
  __syncthreads();
  float acc = b2[j];
  for (int k = 0; k < EQ; ++k) acc += hid[k] * w2[k * EQ + j];
  temb[b * EQ + j] = acc;
}

// im2col for patch embed: row=(b,hy,wx), K = c*256 + p*16 + q
__global__ void k_im2col(const float* __restrict__ x, _Float16* __restrict__ col) {
  int tid = blockIdx.x * blockDim.x + threadIdx.x;
  if (tid >= ROWS * 512) return;
  int row = tid >> 9, kk = tid & 511;
  int c = kk >> 8, p = (kk >> 4) & 15, q = kk & 15;
  int b = row >> 8, l = row & 255;
  int hy = l >> 4, wx = l & 15;
  col[tid] = (_Float16)x[((size_t)((b * 2 + c) * 256) + hy * 16 + p) * 256 + wx * 16 + q];
}

__global__ void k_combine(const float* __restrict__ tok, const float* __restrict__ pos,
                          const float* __restrict__ temb, float* __restrict__ h) {
  int tid = blockIdx.x * blockDim.x + threadIdx.x;
  if (tid >= ROWS * EQ) return;
  int row = tid >> 8, e = tid & 255;
  int b = row >> 8, l = row & 255;
  h[tid] = tok[tid] + pos[l * EQ + e] + temb[b * EQ + e];
}

// ---------------------------------------------------------------- mamba layer

__global__ void k_layernorm(const float* __restrict__ h, const float* __restrict__ g,
                            const float* __restrict__ bb, _Float16* __restrict__ out) {
  __shared__ float red[EQ];
  int row = blockIdx.x, e = threadIdx.x;
  float v = h[(size_t)row * EQ + e];
  red[e] = v; __syncthreads();
  for (int s = 128; s > 0; s >>= 1) { if (e < s) red[e] += red[e + s]; __syncthreads(); }
  float mean = red[0] * (1.f / EQ);
  __syncthreads();
  float d = v - mean;
  red[e] = d * d; __syncthreads();
  for (int s = 128; s > 0; s >>= 1) { if (e < s) red[e] += red[e + s]; __syncthreads(); }
  float var = red[0] * (1.f / EQ);
  out[(size_t)row * EQ + e] = (_Float16)(d * rsqrtf(var + 1e-5f) * g[e] + bb[e]);
}

// causal depthwise conv (DC=4) + SiLU ; u part = xz cols [0,512)
__global__ void k_conv_silu(const float* __restrict__ xz, const float* __restrict__ w,
                            const float* __restrict__ bias, float* __restrict__ uo,
                            _Float16* __restrict__ uo16) {
  int tid = blockIdx.x * blockDim.x + threadIdx.x;
  if (tid >= ROWS * DIQ) return;
  int row = tid >> 9, d = tid & 511;
  int b = row >> 8, l = row & 255;
  float acc = bias[d];
  const float* wd = w + d * 4;
#pragma unroll
  for (int j = 0; j < 4; ++j) {
    int ls = l - 3 + j;
    if (ls >= 0) acc += xz[((size_t)((b << 8) + ls) << 10) + d] * wd[j];
  }
  float s = acc / (1.f + __expf(-acc));
  uo[tid] = s; uo16[tid] = (_Float16)s;
}

// dt = softplus(dbc[:, :16] @ dt_w^T + dt_b)   (K=16 -> VALU)
__global__ void k_dt(const float* __restrict__ dbc, const float* __restrict__ w,
                     const float* __restrict__ bias, float* __restrict__ out) {
  int tid = blockIdx.x * blockDim.x + threadIdx.x;
  if (tid >= ROWS * DIQ) return;
  int row = tid >> 9, d = tid & 511;
  const float* dr = dbc + (size_t)row * 48;
  const float* wr = w + d * 16;
  float acc = bias[d];
#pragma unroll
  for (int r = 0; r < 16; ++r) acc += dr[r] * wr[r];
  out[tid] = (acc > 20.f) ? acc : log1pf(__expf(acc));
}

// selective scan: thread = (b, d) channel, 16-state in registers; B/C staged in LDS
// fused: y = (scan + u*D) * silu(z)  ->  f16 (A operand of out_proj)
__global__ void k_ssm(const float* __restrict__ dbc, const float* __restrict__ dt,
                      const float* __restrict__ u, const float* __restrict__ xz,
                      const float* __restrict__ Alog, const float* __restrict__ Dv,
                      _Float16* __restrict__ y) {
  __shared__ float sBC[32];
  int b = blockIdx.x, d = threadIdx.x;
  float Ac[DSQ], hs[DSQ];
#pragma unroll
  for (int s = 0; s < DSQ; ++s) { Ac[s] = -__expf(Alog[d * DSQ + s]); hs[s] = 0.f; }
  float Dd = Dv[d];
  for (int l = 0; l < LQ; ++l) {
    int row = (b << 8) + l;
    if (d < 32) sBC[d] = dbc[(size_t)row * 48 + 16 + d];   // B then C
    __syncthreads();
    float dv = dt[(size_t)row * DIQ + d];
    float uv = u [(size_t)row * DIQ + d];
    float du = dv * uv;
    float acc = 0.f;
#pragma unroll
    for (int s = 0; s < DSQ; ++s) {
      hs[s] = __expf(dv * Ac[s]) * hs[s] + du * sBC[s];
      acc += hs[s] * sBC[16 + s];
    }
    float zv = xz[((size_t)row << 10) + DIQ + d];
    float yv = (acc + uv * Dd) * (zv / (1.f + __expf(-zv)));
    y[(size_t)row * DIQ + d] = (_Float16)yv;
    __syncthreads();
  }
}

// ---------------------------------------------------------------- decoder

// C[M, 4*O] -> bias + gelu + pixel-shuffle relayout -> A_next[(b,2y+p,2x+q), o] f16
__global__ void k_scatter(const float* __restrict__ Cc, const float* __restrict__ bias,
                          _Float16* __restrict__ Aout, int Hin, int O, long total) {
  long tid = (long)blockIdx.x * blockDim.x + threadIdx.x;
  if (tid >= total) return;
  int N = 4 * O;
  long m = tid / N; int n = (int)(tid - m * N);
  int pq = n / O, o = n - pq * O;
  int p = pq >> 1, q = pq & 1;
  int hw = Hin * Hin;
  int b = (int)(m / hw); int r = (int)(m - (long)b * hw);
  int yy = r / Hin, xx = r - yy * Hin;
  int Hout = Hin * 2;
  float v = Cc[(size_t)m * N + n] + bias[o];
  float g = 0.5f * v * (1.f + erff(v * 0.70710678118654752f));
  Aout[((size_t)(b * Hout + 2 * yy + p) * Hout + (2 * xx + q)) * O + o] = (_Float16)g;
}

// final convT (C=32 -> 1), no gelu, fp32 out
__global__ void k_dec4(const _Float16* __restrict__ A4, const float* __restrict__ w,
                       const float* __restrict__ bias, float* __restrict__ out) {
  int tid = blockIdx.x * blockDim.x + threadIdx.x;
  if (tid >= BQ * 128 * 128) return;
  int b = tid >> 14, r = tid & 16383;
  int yy = r >> 7, xx = r & 127;
  const _Float16* a = A4 + (size_t)tid * 32;
  float av[32];
#pragma unroll
  for (int c = 0; c < 32; ++c) av[c] = (float)a[c];
#pragma unroll
  for (int pq = 0; pq < 4; ++pq) {
    int p = pq >> 1, q = pq & 1;
    float acc = bias[0];
#pragma unroll
    for (int c = 0; c < 32; ++c) acc += av[c] * w[c * 4 + pq];
    out[(size_t)(b * 256 + 2 * yy + p) * 256 + (2 * xx + q)] = acc;
  }
}

// ---------------------------------------------------------------- launch

extern "C" void kernel_launch(void* const* d_in, const int* in_sizes, int n_in,
                              void* d_out, int out_size, void* d_ws, size_t ws_size,
                              hipStream_t stream) {
  const float* x        = (const float*)d_in[0];
  const int*   t        = (const int*)  d_in[1];
  const float* time_w1  = (const float*)d_in[2];
  const float* time_b1  = (const float*)d_in[3];
  const float* time_w2  = (const float*)d_in[4];
  const float* time_b2  = (const float*)d_in[5];
  const float* patch_w  = (const float*)d_in[6];
  const float* patch_b  = (const float*)d_in[7];
  const float* pos      = (const float*)d_in[8];
  const float* ln_g     = (const float*)d_in[9];
  const float* ln_b     = (const float*)d_in[10];
  const float* in_proj  = (const float*)d_in[11];
  const float* conv_w   = (const float*)d_in[12];
  const float* conv_b   = (const float*)d_in[13];
  const float* x_proj   = (const float*)d_in[14];
  const float* dt_w     = (const float*)d_in[15];
  const float* dt_b     = (const float*)d_in[16];
  const float* A_log    = (const float*)d_in[17];
  const float* Dp       = (const float*)d_in[18];
  const float* out_proj = (const float*)d_in[19];
  const float* dec_w1   = (const float*)d_in[20];
  const float* dec_b1   = (const float*)d_in[21];
  const float* dec_w2   = (const float*)d_in[22];
  const float* dec_b2   = (const float*)d_in[23];
  const float* dec_w3   = (const float*)d_in[24];
  const float* dec_b3   = (const float*)d_in[25];
  const float* dec_w4   = (const float*)d_in[26];
  const float* dec_b4   = (const float*)d_in[27];
  float* outp = (float*)d_out;

  // ---- workspace layout (~53 MB total) ----
  char* ws = (char*)d_ws; size_t off = 0;
  auto alloc = [&](size_t bytes) -> char* {
    char* p = ws + off; off += (bytes + 255) & ~(size_t)255; return p;
  };
  float*     temb  = (float*)    alloc((size_t)BQ * EQ * 4);
  float*     hbuf  = (float*)    alloc((size_t)ROWS * EQ * 4);        // 2 MB
  _Float16*  hn16  = (_Float16*) alloc((size_t)ROWS * EQ * 2);        // 1 MB (also dec1 A)
  float*     xz    = (float*)    alloc((size_t)ROWS * 1024 * 4);      // 8 MB (also C2, A4)
  float*     u32   = (float*)    alloc((size_t)ROWS * DIQ * 4);       // 4 MB (also C1)
  _Float16*  u16   = (_Float16*) alloc((size_t)ROWS * DIQ * 2);       // 2 MB
  float*     dbc   = (float*)    alloc((size_t)ROWS * 48 * 4);
  float*     dtb   = (float*)    alloc((size_t)ROWS * DIQ * 4);       // 4 MB (also A3)
  _Float16*  y16   = (_Float16*) alloc((size_t)ROWS * DIQ * 2);       // 2 MB (also A2)
  float*     tok   = (float*)    alloc((size_t)ROWS * EQ * 4);
  _Float16*  col16 = (_Float16*) alloc((size_t)ROWS * 512 * 2);
  _Float16*  pw16  = (_Float16*) alloc((size_t)256 * 512 * 2);
  _Float16*  ip16  = (_Float16*) alloc((size_t)8 * 1024 * 256 * 2);
  _Float16*  xp16  = (_Float16*) alloc((size_t)8 * 48 * 512 * 2);
  _Float16*  op16  = (_Float16*) alloc((size_t)8 * 256 * 512 * 2);
  _Float16*  bt1   = (_Float16*) alloc((size_t)512 * 256 * 2);
  _Float16*  bt2   = (_Float16*) alloc((size_t)256 * 128 * 2);
  _Float16*  bt3   = (_Float16*) alloc((size_t)128 * 64 * 2);
  float*     C3    = (float*)    alloc((size_t)32768 * 128 * 4);      // 16 MB
  // aliases (phase-ordered, never live simultaneously)
  float*     C1 = u32;
  _Float16*  A2 = y16;
  float*     C2 = xz;
  _Float16*  A3 = (_Float16*)dtb;
  _Float16*  A4 = (_Float16*)xz;
  (void)ws_size; (void)in_sizes; (void)n_in; (void)out_size;

  auto nblk = [](long n) { return (unsigned)((n + 255) / 256); };
  auto gemm = [&](const _Float16* A, const _Float16* Bt, const float* bias,
                  const float* add, float* C, int M, int N, int K) {
    if (N % 64 == 0) {
      int tiles = (M >> 4) * (N >> 6);
      k_gemm<4><<<(tiles + 7) / 8, 256, 0, stream>>>(A, Bt, bias, add, C, M, N, K);
    } else {                                   // N = 48 (x_proj)
      int tiles = (M >> 4) * (N / 48);
      k_gemm<3><<<(tiles + 7) / 8, 256, 0, stream>>>(A, Bt, bias, add, C, M, N, K);
    }
  };

  // ---- weight conversions (deterministic each call) ----
  k_cvt<<<nblk(256L*512), 256, 0, stream>>>(patch_w, pw16, 256L*512);
  k_cvt<<<nblk(8L*1024*256), 256, 0, stream>>>(in_proj, ip16, 8L*1024*256);
  k_cvt<<<nblk(8L*48*512), 256, 0, stream>>>(x_proj, xp16, 8L*48*512);
  k_cvt<<<nblk(8L*256*512), 256, 0, stream>>>(out_proj, op16, 8L*256*512);
  k_tdecw<<<nblk(256L*128*4), 256, 0, stream>>>(dec_w1, bt1, 256, 128);
  k_tdecw<<<nblk(128L*64*4), 256, 0, stream>>>(dec_w2, bt2, 128, 64);
  k_tdecw<<<nblk(64L*32*4), 256, 0, stream>>>(dec_w3, bt3, 64, 32);

  // ---- embeddings ----
  k_time_emb<<<BQ, EQ, 0, stream>>>(t, time_w1, time_b1, time_w2, time_b2, temb);
  k_im2col<<<nblk((long)ROWS*512), 256, 0, stream>>>(x, col16);
  gemm(col16, pw16, patch_b, nullptr, tok, ROWS, EQ, 512);
  k_combine<<<nblk((long)ROWS*EQ), 256, 0, stream>>>(tok, pos, temb, hbuf);

  // ---- mamba layers ----
  for (int i = 0; i < 8; ++i) {
    k_layernorm<<<ROWS, EQ, 0, stream>>>(hbuf, ln_g + i*EQ, ln_b + i*EQ, hn16);
    gemm(hn16, ip16 + (size_t)i*1024*256, nullptr, nullptr, xz, ROWS, 1024, 256);
    k_conv_silu<<<nblk((long)ROWS*DIQ), 256, 0, stream>>>(
        xz, conv_w + (size_t)i*DIQ*4, conv_b + (size_t)i*DIQ, u32, u16);
    gemm(u16, xp16 + (size_t)i*48*512, nullptr, nullptr, dbc, ROWS, 48, 512);
    k_dt<<<nblk((long)ROWS*DIQ), 256, 0, stream>>>(
        dbc, dt_w + (size_t)i*DIQ*16, dt_b + (size_t)i*DIQ, dtb);
    k_ssm<<<BQ, DIQ, 0, stream>>>(dbc, dtb, u32, xz,
        A_log + (size_t)i*DIQ*DSQ, Dp + (size_t)i*DIQ, y16);
    gemm(y16, op16 + (size_t)i*256*512, nullptr, hbuf /*residual*/, hbuf, ROWS, EQ, 512);
  }

  // ---- decoder (each convT2 = GEMM with N=4*O + scatter epilogue) ----
  k_cvt<<<nblk((long)ROWS*EQ), 256, 0, stream>>>(hbuf, hn16, (long)ROWS*EQ);
  gemm(hn16, bt1, nullptr, nullptr, C1, 2048, 512, 256);
  k_scatter<<<nblk(2048L*512), 256, 0, stream>>>(C1, dec_b1, A2, 16, 128, 2048L*512);
  gemm(A2, bt2, nullptr, nullptr, C2, 8192, 256, 128);
  k_scatter<<<nblk(8192L*256), 256, 0, stream>>>(C2, dec_b2, A3, 32, 64, 8192L*256);
  gemm(A3, bt3, nullptr, nullptr, C3, 32768, 128, 64);
  k_scatter<<<nblk(32768L*128), 256, 0, stream>>>(C3, dec_b3, A4, 64, 32, 32768L*128);
  k_dec4<<<nblk((long)BQ*128*128), 256, 0, stream>>>(A4, dec_w4, dec_b4, outp);
}
